// APPNPModel_82566451298751
// MI455X (gfx1250) — matrix operand order repaired and checked by reading, not compiled, not run on hardware
//
#include <hip/hip_runtime.h>

#define F_IN    512
#define F_OUT   64
#define K_STEPS 50
#define ALPHA_F 0.1f

typedef __attribute__((ext_vector_type(16))) _Float16 v16h;
typedef __attribute__((ext_vector_type(8)))  float    v8f;

// ---------- W convert+transpose: Wt[n][k] = (f16) W[k][n] ----------
__global__ void appnp_wconv(const float* __restrict__ W, _Float16* __restrict__ Wt) {
  int idx = blockIdx.x * blockDim.x + threadIdx.x;
  if (idx < F_IN * F_OUT) {
    int k = idx >> 6;   // / F_OUT
    int n = idx & 63;   // % F_OUT
    Wt[(size_t)n * F_IN + k] = (_Float16)W[idx];
  }
}

// ---------- h = relu(x @ W + b): B staged in LDS, 32 rows per wave ----------
__global__ void __launch_bounds__(256)
appnp_gemm_relu(const float* __restrict__ x,
                const _Float16* __restrict__ Wt,   // [64][512] col-major (f16)
                const float* __restrict__ bias,
                float* __restrict__ h, int N) {
  // Stage the whole weight matrix (64KB f16) in LDS once per block.
  __shared__ _Float16 Wl[F_OUT * F_IN];
  {
    const int nvec = (F_OUT * F_IN) / 8;     // 16B vectors
    for (int i = threadIdx.x; i < nvec; i += blockDim.x)
      ((float4*)Wl)[i] = ((const float4*)Wt)[i];
  }
  __syncthreads();

  const int wavesPerBlock = blockDim.x >> 5;
  const int wave = blockIdx.x * wavesPerBlock + ((int)threadIdx.x >> 5);
  const int lane = (int)threadIdx.x & 31;
  const int row0 = wave << 5;                // 32 rows per wave (2 M-tiles)
  if (row0 >= N) return;

  const int m  = lane & 15;   // A: row within tile, B/D: col within tile
  const int kh = lane >> 4;   // K-half selector: element s holds k = 16*kh + s (contiguous order)
  const float* xrow0 = x + (size_t)(row0 + m) * F_IN;
  const float* xrow1 = x + (size_t)(row0 + 16 + m) * F_IN;

  v8f acc[2][4] = {};

  for (int k0 = 0; k0 < F_IN; k0 += 32) {
    // Two A fragments (rows row0..+15 and row0+16..+31), f32 -> f16
    v16h a0, a1;
    const float* p00 = xrow0 + k0 + kh * 16;
    const float* p01 = xrow0 + k0 + kh * 16 + 8;
    const float* p10 = xrow1 + k0 + kh * 16;
    const float* p11 = xrow1 + k0 + kh * 16 + 8;
#pragma unroll
    for (int j = 0; j < 8; ++j) {
      a0[j]     = (_Float16)p00[j];
      a0[8 + j] = (_Float16)p01[j];
      a1[j]     = (_Float16)p10[j];
      a1[8 + j] = (_Float16)p11[j];
    }
#pragma unroll
    for (int nt = 0; nt < 4; ++nt) {
      // B fragment from LDS: lane holds col nt*16+m, K = k0+kh*16..+15 (32B contig)
      v16h bf = *(const v16h*)&Wl[(size_t)(nt * 16 + m) * F_IN + k0 + kh * 16];
      acc[0][nt] = __builtin_amdgcn_wmma_f32_16x16x32_f16(
          false, a0, false, bf, (short)0, acc[0][nt], false, false);
      acc[1][nt] = __builtin_amdgcn_wmma_f32_16x16x32_f16(
          false, a1, false, bf, (short)0, acc[1][nt], false, false);
    }
  }

  // D layout: lanes 0-15 -> M=r, lanes 16-31 -> M=8+r; N = lane&15
#pragma unroll
  for (int rt = 0; rt < 2; ++rt) {
#pragma unroll
    for (int nt = 0; nt < 4; ++nt) {
      const int col = nt * 16 + m;
      const float bv = bias[col];
      float* hp = h + (size_t)(row0 + rt * 16 + kh * 8) * F_OUT + col;
#pragma unroll
      for (int r = 0; r < 8; ++r) {
        float v = acc[rt][nt][r] + bv;
        hp[(size_t)r * F_OUT] = v > 0.f ? v : 0.f;
      }
    }
  }
}

// ---------- degree (self-loop folded in as init=1) ----------
__global__ void appnp_deg_init(float* __restrict__ deg, int N) {
  int i = blockIdx.x * blockDim.x + threadIdx.x;
  if (i < N) deg[i] = 1.0f;
}

__global__ void appnp_deg_edges(const long long* __restrict__ dst,
                                float* __restrict__ deg, int E) {
  int e = blockIdx.x * blockDim.x + threadIdx.x;
  if (e < E) atomicAdd(&deg[dst[e]], 1.0f);
}

__global__ void appnp_dinv(const float* __restrict__ deg,
                           float* __restrict__ dinv, int N) {
  int i = blockIdx.x * blockDim.x + threadIdx.x;
  if (i < N) dinv[i] = rsqrtf(deg[i]);  // deg >= 1 always
}

__global__ void appnp_norm(const long long* __restrict__ src,
                           const long long* __restrict__ dst,
                           const float* __restrict__ dinv,
                           float* __restrict__ norm, int E) {
  int e = blockIdx.x * blockDim.x + threadIdx.x;
  if (e < E) norm[e] = dinv[src[e]] * dinv[dst[e]];
}

// ---------- one-time: agg = dinv^2 * carry (self-loop term) ----------
__global__ void appnp_self_init(const float* __restrict__ carry,
                                const float* __restrict__ dinv,
                                float4* __restrict__ agg4, int n4 /* N*16 */) {
  int idx = blockIdx.x * blockDim.x + threadIdx.x;
  if (idx < n4) {
    int node = idx >> 4;                // 16 float4 per node row
    float d = dinv[node];
    float s = d * d;
    const float4 c = ((const float4*)carry)[idx];
    float4 o; o.x = s * c.x; o.y = s * c.y; o.z = s * c.z; o.w = s * c.w;
    agg4[idx] = o;
  }
}

// ---------- per-step: edge scatter, one wave32 per edge, 2 feats/lane ----------
__global__ void appnp_edge_prop(const long long* __restrict__ src,
                                const long long* __restrict__ dst,
                                const float* __restrict__ norm,
                                const float* __restrict__ carry,
                                float* __restrict__ agg, int E) {
  int wave = blockIdx.x * (blockDim.x >> 5) + ((int)threadIdx.x >> 5);
  int lane = (int)threadIdx.x & 31;
  if (wave >= E) return;

  long long s = src[wave];
  long long d = dst[wave];
  float nv = norm[wave];

  // prefetch next edge's gather row (gfx1250 global_prefetch_b8)
  int nxt = wave + (blockDim.x >> 5);
  if (nxt < E) __builtin_prefetch(carry + (size_t)src[nxt] * F_OUT, 0, 0);

  const float2 c = ((const float2*)(carry + (size_t)s * F_OUT))[lane];
  float* ap = agg + (size_t)d * F_OUT + 2 * lane;
  atomicAdd(&ap[0], nv * c.x);
  atomicAdd(&ap[1], nv * c.y);
}

// ---------- fused: carry = 0.9*agg + 0.1*h ; agg_next = dinv^2 * carry ----------
__global__ void appnp_combine_init(const float4* __restrict__ agg4,
                                   const float4* __restrict__ h4,
                                   const float* __restrict__ dinv,
                                   float4* __restrict__ carry4,
                                   float4* __restrict__ aggn4, int n4) {
  int idx = blockIdx.x * blockDim.x + threadIdx.x;
  if (idx < n4) {
    int node = idx >> 4;
    float d = dinv[node];
    float s = d * d;
    float4 a = agg4[idx], hh = h4[idx];
    float4 c;
    c.x = (1.0f - ALPHA_F) * a.x + ALPHA_F * hh.x;
    c.y = (1.0f - ALPHA_F) * a.y + ALPHA_F * hh.y;
    c.z = (1.0f - ALPHA_F) * a.z + ALPHA_F * hh.z;
    c.w = (1.0f - ALPHA_F) * a.w + ALPHA_F * hh.w;
    carry4[idx] = c;
    float4 o; o.x = s * c.x; o.y = s * c.y; o.z = s * c.z; o.w = s * c.w;
    aggn4[idx] = o;
  }
}

// ---------- final step: out = 0.9*agg + 0.1*h ----------
__global__ void appnp_combine(const float4* __restrict__ agg4,
                              const float4* __restrict__ h4,
                              float4* __restrict__ out4, int n4) {
  int idx = blockIdx.x * blockDim.x + threadIdx.x;
  if (idx < n4) {
    float4 a = agg4[idx], hh = h4[idx];
    float4 o;
    o.x = (1.0f - ALPHA_F) * a.x + ALPHA_F * hh.x;
    o.y = (1.0f - ALPHA_F) * a.y + ALPHA_F * hh.y;
    o.z = (1.0f - ALPHA_F) * a.z + ALPHA_F * hh.z;
    o.w = (1.0f - ALPHA_F) * a.w + ALPHA_F * hh.w;
    out4[idx] = o;
  }
}

static inline size_t align256(size_t v) { return (v + 255) & ~(size_t)255; }

extern "C" void kernel_launch(void* const* d_in, const int* in_sizes, int n_in,
                              void* d_out, int out_size, void* d_ws, size_t ws_size,
                              hipStream_t stream) {
  const float*     x  = (const float*)d_in[0];
  const long long* ei = (const long long*)d_in[1];   // int64 [2, E]
  const float*     W  = (const float*)d_in[2];
  const float*     b  = (const float*)d_in[3];

  const int N = in_sizes[0] / F_IN;   // 100000
  const int E = in_sizes[1] / 2;      // 1600000
  const long long* src = ei;
  const long long* dst = ei + E;

  // ---- workspace carve-out ----
  char* ws = (char*)d_ws;
  size_t off = 0;
  _Float16* Wt   = (_Float16*)(ws + off); off += align256((size_t)F_IN * F_OUT * sizeof(_Float16));
  float*    h    = (float*)   (ws + off); off += align256((size_t)N * F_OUT * sizeof(float));
  float*    deg  = (float*)   (ws + off); off += align256((size_t)N * sizeof(float));
  float*    dinv = (float*)   (ws + off); off += align256((size_t)N * sizeof(float));
  float*    nrm  = (float*)   (ws + off); off += align256((size_t)E * sizeof(float));
  float*    aggA = (float*)   (ws + off); off += align256((size_t)N * F_OUT * sizeof(float));
  float*    aggB = (float*)   (ws + off); off += align256((size_t)N * F_OUT * sizeof(float));
  float*    carA = (float*)   (ws + off); off += align256((size_t)N * F_OUT * sizeof(float));
  float*    outp = (float*)d_out;
  (void)ws_size; (void)n_in; (void)out_size;

  const int B = 256;

  // 1) W -> f16, transposed
  appnp_wconv<<<(F_IN * F_OUT + B - 1) / B, B, 0, stream>>>(W, Wt);

  // 2) h = relu(x @ W + b)   (WMMA: 32 rows/wave, 8 waves/block, W in LDS)
  {
    int waves = (N + 31) / 32;
    int blocks = (waves + 7) / 8;
    appnp_gemm_relu<<<blocks, B, 0, stream>>>(x, Wt, b, h, N);
  }

  // 3) degrees + dinv + per-edge norm
  appnp_deg_init<<<(N + B - 1) / B, B, 0, stream>>>(deg, N);
  appnp_deg_edges<<<(E + B - 1) / B, B, 0, stream>>>(dst, deg, E);
  appnp_dinv<<<(N + B - 1) / B, B, 0, stream>>>(deg, dinv, N);
  appnp_norm<<<(E + B - 1) / B, B, 0, stream>>>(src, dst, dinv, nrm, E);

  // 4) K propagation steps; agg double-buffered, carry ping-pongs carA <-> d_out
  const int n4 = N * (F_OUT / 4);
  const int nBlocks4   = (n4 + B - 1) / B;
  const int edgeBlocks = (E + 7) / 8;   // one wave32 per edge, 8 waves/block

  appnp_self_init<<<nBlocks4, B, 0, stream>>>(h, dinv, (float4*)aggA, n4);

  const float* prev = h;
  float* aggCur = aggA;
  float* aggNxt = aggB;
  for (int k = 1; k <= K_STEPS; ++k) {
    appnp_edge_prop<<<edgeBlocks, B, 0, stream>>>(src, dst, nrm, prev, aggCur, E);
    if (k < K_STEPS) {
      float* next = (k & 1) ? carA : outp;
      appnp_combine_init<<<nBlocks4, B, 0, stream>>>(
          (const float4*)aggCur, (const float4*)h, dinv,
          (float4*)next, (float4*)aggNxt, n4);
      prev = next;
      float* t = aggCur; aggCur = aggNxt; aggNxt = t;
    } else {
      appnp_combine<<<nBlocks4, B, 0, stream>>>(
          (const float4*)aggCur, (const float4*)h, (float4*)outp, n4);
    }
  }
}
